// RBMSymm_Expand_22874995819122
// MI455X (gfx1250) — matrix operand order repaired and checked by reading, not compiled, not run on hardware
//
#include <hip/hip_runtime.h>

typedef __attribute__((ext_vector_type(16))) __bf16 v16bf;
typedef __attribute__((ext_vector_type(8)))  float  v8f;

#define N_SITES 1024
#define ALPHA   16
#define LN2F    0.69314718055994530942f

// ---- bf16 helpers (bit-level, round-to-nearest-even) ----
__device__ __forceinline__ unsigned bf16_rne(float f) {
  unsigned u = __float_as_uint(f);
  return (u + 0x7FFFu + ((u >> 16) & 1u)) >> 16;
}
__device__ __forceinline__ float bf16_as_f32(unsigned h) {
  return __uint_as_float(h << 16);
}
__device__ __forceinline__ void bf16_split(float f, unsigned& h, unsigned& l) {
  h = bf16_rne(f);
  l = bf16_rne(f - bf16_as_f32(h));
}
__device__ __forceinline__ unsigned pack_hi2(float f0, float f1) {
  return bf16_rne(f0) | (bf16_rne(f1) << 16);
}
__device__ __forceinline__ unsigned pack_lo2(float f0, float f1) {
  float r0 = f0 - bf16_as_f32(bf16_rne(f0));
  float r1 = f1 - bf16_as_f32(bf16_rne(f1));
  return bf16_rne(r0) | (bf16_rne(r1) << 16);
}

union BfFrag { v16bf v; unsigned u[8]; uint2 d[4]; uint4 q[2]; };

// logcosh(x) + ln2  (the -ln2 constant is folded into the init kernel)
__device__ __forceinline__ float logcosh_noln2(float v) {
  float a = fabsf(v);
  return a + __logf(1.0f + __expf(-2.0f * a));
}

// ----------------------------------------------------------------------------
// Pre-pass 1: out[b] = v_bias * sum_i x[b,i] - (alpha*n_sites)*ln2
// ----------------------------------------------------------------------------
__global__ __launch_bounds__(256) void rbm_init_out(
    const float* __restrict__ x, const float* __restrict__ v_bias,
    float* __restrict__ out) {
  __shared__ float red[8];
  const int b = blockIdx.x;
  const int tid = threadIdx.x;
  float s = 0.f;
  for (int i = tid; i < N_SITES; i += 256) s += x[(size_t)b * N_SITES + i];
  #pragma unroll
  for (int m = 1; m < 32; m <<= 1) s += __shfl_xor(s, m, 32);
  if ((tid & 31) == 0) red[tid >> 5] = s;
  __syncthreads();
  if (tid == 0) {
    float tot = 0.f;
    #pragma unroll
    for (int w = 0; w < 8; ++w) tot += red[w];
    out[b] = v_bias[0] * tot - (float)(ALPHA * N_SITES) * LN2F;
  }
}

// ----------------------------------------------------------------------------
// Pre-pass 2: split x into bf16 hi/lo, pair-packed along K (A-fragment order).
// xh/xl: [batch][512] dwords, dword m = pack(x[b,2m], x[b,2m+1]).
// ----------------------------------------------------------------------------
__global__ __launch_bounds__(256) void rbm_cvt_x(
    const float* __restrict__ x, unsigned* __restrict__ xh,
    unsigned* __restrict__ xl) {
  const size_t b = blockIdx.x;
  for (int m = threadIdx.x; m < N_SITES / 2; m += 256) {
    const float2 f = *(const float2*)(x + b * N_SITES + 2 * m);
    xh[b * (N_SITES / 2) + m] = pack_hi2(f.x, f.y);
    xl[b * (N_SITES / 2) + m] = pack_lo2(f.x, f.y);
  }
}

// ----------------------------------------------------------------------------
// Main: split-bf16 WMMA GEMM over the circulant-expanded kernel + logcosh.
// block = 256 threads (8 wave32); tile = 256 batch rows x 64 k-values x 1 j.
// Each wave: 32 rows (2 A sets) x 4 N tiles = 8 C fragments, 24 WMMA / K-step.
// ----------------------------------------------------------------------------
__global__ __launch_bounds__(256) void rbm_symm_wmma(
    const unsigned* __restrict__ xh, const unsigned* __restrict__ xl,
    const float* __restrict__ W, const float* __restrict__ bsymm,
    float* __restrict__ out) {
  // Shifted column j of W as bf16 hi/lo pair-packs. Planes are split by site
  // parity p and ADDITIONALLY stored dword-shifted by q in {0,1}:
  //   plane(p,q)[n] = pack(w[2(n+q)+p], w[2(n+q)+p+1])
  // so every lane's 8 B-fragment dwords start at an EVEN dword -> uint2 loads
  // -> ds_load_2addr_b64 (16B per DS issue, no deinterleave moves).
  __shared__ unsigned bhP[4 * N_SITES];  // 16 KB
  __shared__ unsigned blP[4 * N_SITES];  // 16 KB

  const int j      = blockIdx.z;
  const int k_base = blockIdx.y * 64;
  const int b0     = blockIdx.x * 256;
  const int tid    = threadIdx.x;
  const int lane   = tid & 31;
  const int wave   = tid >> 5;
  const int kh     = lane >> 4;   // lane half (0/1)
  const int ln     = lane & 15;   // lane within half

  for (int m = tid; m < N_SITES; m += 256) {
    unsigned h0, h1, h2, h3, h4, l0, l1, l2, l3, l4;
    bf16_split(W[((2 * m)     & (N_SITES - 1)) * ALPHA + j], h0, l0);
    bf16_split(W[((2 * m + 1) & (N_SITES - 1)) * ALPHA + j], h1, l1);
    bf16_split(W[((2 * m + 2) & (N_SITES - 1)) * ALPHA + j], h2, l2);
    bf16_split(W[((2 * m + 3) & (N_SITES - 1)) * ALPHA + j], h3, l3);
    bf16_split(W[((2 * m + 4) & (N_SITES - 1)) * ALPHA + j], h4, l4);
    bhP[0 * N_SITES + m] = h0 | (h1 << 16);  // p=0,q=0
    bhP[1 * N_SITES + m] = h2 | (h3 << 16);  // p=0,q=1
    bhP[2 * N_SITES + m] = h1 | (h2 << 16);  // p=1,q=0
    bhP[3 * N_SITES + m] = h3 | (h4 << 16);  // p=1,q=1
    blP[0 * N_SITES + m] = l0 | (l1 << 16);
    blP[1 * N_SITES + m] = l2 | (l3 << 16);
    blP[2 * N_SITES + m] = l1 | (l2 << 16);
    blP[3 * N_SITES + m] = l3 | (l4 << 16);
  }
  __syncthreads();

  // Per-lane B plane selection (site parity p, dword parity q are loop-invariant)
  const int s0    = kh * 16 + k_base + ln;  // site index at i0=0, t=0
  const int p     = s0 & 1;
  const int m0b   = s0 >> 1;
  const int q     = m0b & 1;
  const int base0 = m0b - q;                // even
  const unsigned* __restrict__ Ph = bhP + (p * 2 + q) * N_SITES;
  const unsigned* __restrict__ Pl = blP + (p * 2 + q) * N_SITES;

  // Per-lane A source rows (16-bit A layout: M = ln for both lane halves).
  const size_t row0 = (size_t)(b0 + wave * 32 + ln) * (N_SITES / 2);
  const uint4* __restrict__ ah4_0 = (const uint4*)(xh + row0);
  const uint4* __restrict__ al4_0 = (const uint4*)(xl + row0);
  const uint4* __restrict__ ah4_1 = (const uint4*)(xh + row0 + 16 * (N_SITES / 2));
  const uint4* __restrict__ al4_1 = (const uint4*)(xl + row0 + 16 * (N_SITES / 2));

  v8f acc[2][4];
  #pragma unroll
  for (int s = 0; s < 2; ++s)
    #pragma unroll
    for (int t = 0; t < 4; ++t)
      acc[s][t] = (v8f){0.f, 0.f, 0.f, 0.f, 0.f, 0.f, 0.f, 0.f};

  for (int i0 = 0; i0 < N_SITES; i0 += 32) {
    // A fragments: dwords {kh*8 + 0..7} of this K window
    // => two 16B-aligned b128 loads per precision per row set, zero VALU.
    BfFrag ah0, al0, ah1, al1;
    const int q0 = (i0 >> 3) + 2 * kh;
    ah0.q[0] = ah4_0[q0];  ah0.q[1] = ah4_0[q0 + 1];
    al0.q[0] = al4_0[q0];  al0.q[1] = al4_0[q0 + 1];
    ah1.q[0] = ah4_1[q0];  ah1.q[1] = ah4_1[q0 + 1];
    al1.q[0] = al4_1[q0];  al1.q[1] = al4_1[q0 + 1];

    const int nb = base0 + (i0 >> 1);
    #pragma unroll
    for (int t = 0; t < 4; ++t) {
      // B fragment: 8 consecutive dwords at even base -> 4 uint2 LDS loads.
      BfFrag bh, bl;
      const uint2* __restrict__ bph = (const uint2*)(Ph + nb + 8 * t);
      const uint2* __restrict__ bpl = (const uint2*)(Pl + nb + 8 * t);
      #pragma unroll
      for (int v = 0; v < 4; ++v) bh.d[v] = bph[v];
      #pragma unroll
      for (int v = 0; v < 4; ++v) bl.d[v] = bpl[v];

      // split-bf16: x*w ~= xh*wh + xl*wh + xh*wl; 6 WMMAs round-robin 2 accs.
      acc[0][t] = __builtin_amdgcn_wmma_f32_16x16x32_bf16(
                      false, ah0.v, false, bh.v, (short)0, acc[0][t], false, false);
      acc[1][t] = __builtin_amdgcn_wmma_f32_16x16x32_bf16(
                      false, ah1.v, false, bh.v, (short)0, acc[1][t], false, false);
      acc[0][t] = __builtin_amdgcn_wmma_f32_16x16x32_bf16(
                      false, al0.v, false, bh.v, (short)0, acc[0][t], false, false);
      acc[1][t] = __builtin_amdgcn_wmma_f32_16x16x32_bf16(
                      false, al1.v, false, bh.v, (short)0, acc[1][t], false, false);
      acc[0][t] = __builtin_amdgcn_wmma_f32_16x16x32_bf16(
                      false, ah0.v, false, bl.v, (short)0, acc[0][t], false, false);
      acc[1][t] = __builtin_amdgcn_wmma_f32_16x16x32_bf16(
                      false, ah1.v, false, bl.v, (short)0, acc[1][t], false, false);
    }
  }

  // Epilogue: bias + logcosh, reduce the 16 N-values of each C row across the
  // owning 16-lane half (C layout: VGPR r -> M = r + 8*kh, N = ln).
  const float bj = bsymm[j];
  #pragma unroll
  for (int s = 0; s < 2; ++s) {
    float rsum[8];
    #pragma unroll
    for (int r = 0; r < 8; ++r) {
      float v = 0.f;
      #pragma unroll
      for (int t = 0; t < 4; ++t) v += logcosh_noln2(acc[s][t][r] + bj);
      #pragma unroll
      for (int m = 1; m <= 8; m <<= 1) v += __shfl_xor(v, m, 32);
      rsum[r] = v;
    }
    if (ln == 0) {
      const int brow = b0 + wave * 32 + s * 16 + kh * 8;
      #pragma unroll
      for (int r = 0; r < 8; ++r) atomicAdd(&out[brow + r], rsum[r]);
    }
  }
}

// ----------------------------------------------------------------------------
extern "C" void kernel_launch(void* const* d_in, const int* in_sizes, int n_in,
                              void* d_out, int out_size, void* d_ws, size_t ws_size,
                              hipStream_t stream) {
  const float* x  = (const float*)d_in[0];  // [batch, 1024]
  const float* W  = (const float*)d_in[1];  // [1024, 16]
  const float* bs = (const float*)d_in[2];  // [16]
  const float* vb = (const float*)d_in[3];  // [1]
  // d_in[4] = perms (cyclic shift by construction; structure used analytically)
  float* out = (float*)d_out;               // [batch]

  const int batch = in_sizes[0] / N_SITES;

  unsigned* xh = (unsigned*)d_ws;                       // batch*512 dwords
  unsigned* xl = xh + (size_t)batch * (N_SITES / 2);    // batch*512 dwords

  rbm_init_out<<<batch, 256, 0, stream>>>(x, vb, out);
  rbm_cvt_x<<<batch, 256, 0, stream>>>(x, xh, xl);

  dim3 grid(batch / 256, N_SITES / 64, ALPHA);
  rbm_symm_wmma<<<grid, 256, 0, stream>>>(xh, xl, W, bs, out);
}
